// CausalSelfAttention_85718957293707
// MI455X (gfx1250) — matrix operand order repaired and checked by reading, not compiled, not run on hardware
//
#include <hip/hip_runtime.h>

// ---------- types ----------
typedef __attribute__((ext_vector_type(16))) __bf16 v16bf;
typedef __attribute__((ext_vector_type(8)))  float  v8f;
typedef unsigned short u16;
typedef unsigned int   u32;

#define D_MODEL 1024
#define N_HEADS 16
#define DK      64
#define BB      4
#define TT      2048
#define MROWS   (BB*TT)        // 8192 rows of x / q / k / v / ctx

// float -> bf16, round-to-nearest-even
__device__ __forceinline__ u16 f2bf(float f) {
    u32 u = __float_as_uint(f);
    u32 r = (u + 0x7FFFu + ((u >> 16) & 1u)) >> 16;
    return (u16)r;
}

union FragU {
    v16bf  v;
    float4 f4[2];
    u16    u[16];
};

__device__ __forceinline__ v8f wmma_bf16(const FragU& a, const FragU& b, v8f c) {
    // D = A(16x32 bf16) x B(32x16 bf16) + C(16x16 f32)
    return __builtin_amdgcn_wmma_f32_16x16x32_bf16(
        false, a.v, false, b.v, (short)0, c, false, false);
}

// Async global->LDS copy, 16 bytes, tracked by ASYNCcnt (gfx1250).
// lds_byte: LDS byte address (low 32 bits of flat shared address);
// base: uniform 64-bit base (SGPR); goff: per-lane byte offset (VGPR).
__device__ __forceinline__ void async_b128(u32 lds_byte, const void* base, u32 goff) {
    asm volatile("global_load_async_to_lds_b128 %0, %1, %2"
                 :: "v"(lds_byte), "v"(goff), "s"(base)
                 : "memory");
}
__device__ __forceinline__ void wait_async0() {
    asm volatile("s_wait_asynccnt 0x0" ::: "memory");
}

// ---------- f32 -> bf16 conversion (vectorized x4) ----------
__global__ __launch_bounds__(256)
void cvt_f32_bf16(const float* __restrict__ s, u16* __restrict__ d, int n4) {
    int i = blockIdx.x * blockDim.x + threadIdx.x;
    if (i >= n4) return;
    float4 v = ((const float4*)s)[i];
    ushort4 o;
    o.x = f2bf(v.x); o.y = f2bf(v.y); o.z = f2bf(v.z); o.w = f2bf(v.w);
    ((ushort4*)d)[i] = o;
}

// ---------- GEMM: Y = X @ W^T + bias, async-LDS double-buffered ----------
// X: [MROWS, D_MODEL] bf16 row-major, W: [D_MODEL, D_MODEL] bf16 row-major.
// OUT_MODE: 0 = bf16 row-major, 1 = bf16 V-transposed [b,h,d,t], 2 = f32 row-major
// WG = 8 waves (2 m-groups x 4 n-groups); wave tile 32x64; WG tile 64x256; BK=64.
// LDS: double-buffered A(64x64) + B(256x64) bf16 = 80KB.
#define GA_HALVES (64*64)      // 4096
#define GB_HALVES (256*64)     // 16384

// one pipeline stage: each of 256 threads issues 2 A + 8 B async b128 copies
__device__ __forceinline__ void gemm_stage_async(const u16* X, const u16* W,
                                                 int mblk, int nblk, int kb,
                                                 u32 lds_a, u32 lds_b, int tid) {
    // A tile: 64 rows x 64 halves; thread t -> row t/4, halves [(t%4)*16, +16)
    const int m = tid >> 2;
    const int c = (tid & 3) * 16;
    const u32 ga = ((u32)(mblk + m) * D_MODEL + (u32)kb + (u32)c) * 2u;
    const u32 la = lds_a + ((u32)m * 64 + (u32)c) * 2u;
    async_b128(la,      X, ga);
    async_b128(la + 16, X, ga + 16);
    // B tile: 256 rows x 64 halves; thread t -> whole row t (128B = 8 x b128)
    const u32 gb = ((u32)(nblk + tid) * D_MODEL + (u32)kb) * 2u;
    const u32 lb = lds_b + (u32)tid * 128u;
#pragma unroll
    for (int s = 0; s < 8; ++s)
        async_b128(lb + s * 16, W, gb + s * 16);
}

template<int OUT_MODE>
__global__ __launch_bounds__(256)
void gemm_xwT(const u16* __restrict__ X, const u16* __restrict__ W,
              const float* __restrict__ bias, void* __restrict__ Yv) {
    __shared__ __align__(16) u16 sbuf[2 * GA_HALVES + 2 * GB_HALVES];  // 80 KB

    const int tid  = threadIdx.x;
    const int lane = tid & 31;
    const int wave = tid >> 5;
    const int lo = lane & 15, hi = lane >> 4;
    const int wm = wave >> 2, wn = wave & 3;
    const int mblk = blockIdx.y * 64;
    const int nblk = blockIdx.x * 256;

    // LDS base byte address: flat shared address truncated to 32 bits = LDS offset.
    // (scalar runtime expression only -- no LDS-derived aggregate initializers!)
    const u32 sbase = (u32)(size_t)(void*)sbuf;

    v8f acc[2][4];
    const v8f zero = {0.f,0.f,0.f,0.f,0.f,0.f,0.f,0.f};
#pragma unroll
    for (int i = 0; i < 2; ++i)
#pragma unroll
        for (int j = 0; j < 4; ++j) acc[i][j] = zero;

    // prologue: stage 0 into buffer 0
    gemm_stage_async(X, W, mblk, nblk, 0,
                     sbase, sbase + 4u * GA_HALVES, tid);
    wait_async0();
    __syncthreads();

    const int NIT = D_MODEL / 64;   // 16
    for (int it = 0; it < NIT; ++it) {
        const int cur = it & 1;
        const u32 ldsA_nxt = sbase + (u32)(cur ^ 1) * (GA_HALVES * 2u);
        const u32 ldsB_nxt = sbase + 4u * GA_HALVES + (u32)(cur ^ 1) * (GB_HALVES * 2u);
        if (it + 1 < NIT)   // prefetch next stage into the other buffer (async engine)
            gemm_stage_async(X, W, mblk, nblk, (it + 1) * 64,
                             ldsA_nxt, ldsB_nxt, tid);

        const u16* As = sbuf + (size_t)cur * GA_HALVES;
        const u16* Bs = sbuf + 2 * GA_HALVES + (size_t)cur * GB_HALVES;
#pragma unroll
        for (int ks = 0; ks < 64; ks += 32) {
            FragU a0, a1, bfr[4];
            const u16* ar0 = As + (wm * 32 + lo) * 64 + ks;
            const u16* ar1 = ar0 + 16 * 64;
            // A-frag: element e -> k = (e/8)*16 + hi*8 + e%8
            a0.f4[0] = *(const float4*)(ar0 +      hi * 8);
            a0.f4[1] = *(const float4*)(ar0 + 16 + hi * 8);
            a1.f4[0] = *(const float4*)(ar1 +      hi * 8);
            a1.f4[1] = *(const float4*)(ar1 + 16 + hi * 8);
            // B-frag: element e -> k = (e/8)*16 + hi*8 + e%8, N = lo
#pragma unroll
            for (int j = 0; j < 4; ++j) {
                const u16* br = Bs + (wn * 64 + j * 16 + lo) * 64 + ks;
                bfr[j].f4[0] = *(const float4*)(br +      hi * 8);
                bfr[j].f4[1] = *(const float4*)(br + 16 + hi * 8);
            }
#pragma unroll
            for (int j = 0; j < 4; ++j) {
                acc[0][j] = wmma_bf16(a0, bfr[j], acc[0][j]);
                acc[1][j] = wmma_bf16(a1, bfr[j], acc[1][j]);
            }
        }
        wait_async0();      // own prefetch landed
        __syncthreads();    // whole WG's prefetch landed; readers of 'cur' done
    }

    // store: C-layout element r -> row = mbase + i*16 + r + 8*hi, col = nbase + j*16 + lo
    const int mbase = mblk + wm * 32;
    const int nbase = nblk + wn * 64;
#pragma unroll
    for (int i = 0; i < 2; ++i) {
#pragma unroll
        for (int j = 0; j < 4; ++j) {
            const int col = nbase + j * 16 + lo;
            const float bv = bias[col];
#pragma unroll
            for (int r = 0; r < 8; ++r) {
                const int row = mbase + i * 16 + r + 8 * hi;
                const float val = acc[i][j][r] + bv;
                if (OUT_MODE == 0) {
                    ((u16*)Yv)[(size_t)row * D_MODEL + col] = f2bf(val);
                } else if (OUT_MODE == 1) {
                    // V stored transposed: Vt[(b*H + h)*DK + d][t], t contiguous
                    const int bb = row >> 11, t = row & (TT - 1);
                    const int hh = col >> 6,  d = col & (DK - 1);
                    ((u16*)Yv)[((size_t)((bb * N_HEADS + hh) * DK + d) << 11) + t] = f2bf(val);
                } else {
                    ((float*)Yv)[(size_t)row * D_MODEL + col] = val;
                }
            }
        }
    }
}

// ---------- flash attention ----------
// 1 wave handles one (b, h, 16-query tile). 8 waves / block. grid = B*H*(T/16)/8.
__global__ __launch_bounds__(256)
void attn_kernel(const u16* __restrict__ Q, const u16* __restrict__ K,
                 const u16* __restrict__ Vt, u16* __restrict__ Ctx) {
    __shared__ __align__(16) u16 pbuf[8][16 * 32];   // per-wave P tile (16x32 bf16)

    const int lane = threadIdx.x & 31;
    const int wave = threadIdx.x >> 5;
    const int lo = lane & 15, hi = lane >> 4;
    const int wid = blockIdx.x * 8 + wave;
    const int bh  = wid >> 7;            // (b*H + h)
    const int qb  = (wid & 127) << 4;    // query tile base
    const int b   = bh >> 4, h = bh & 15;

    // Q A-fragments for d=0..31 and d=32..63
    const u16* qrow = Q + (size_t)(b * TT + qb + lo) * D_MODEL + h * DK;
    FragU qf[2];
    qf[0].f4[0] = *(const float4*)(qrow +      hi*8);
    qf[0].f4[1] = *(const float4*)(qrow + 16 + hi*8);
    qf[1].f4[0] = *(const float4*)(qrow + 32 + hi*8);
    qf[1].f4[1] = *(const float4*)(qrow + 48 + hi*8);

    const v8f zero = {0.f,0.f,0.f,0.f,0.f,0.f,0.f,0.f};
    v8f o[4] = {zero, zero, zero, zero};
    float mr[8], lr[8];
#pragma unroll
    for (int r = 0; r < 8; ++r) { mr[r] = -1e30f; lr[r] = 0.f; }

    const int nch = ((qb + 15) >> 5) + 1;   // causal: key chunks of 32
    for (int ch = 0; ch < nch; ++ch) {
        const int kb = ch * 32;
        // S = Q @ K^T for two 16-key tiles (contraction over d = 64 -> 2 WMMAs each)
        v8f s[2];
#pragma unroll
        for (int t = 0; t < 2; ++t) {
            const u16* krow = K + (size_t)(b * TT + kb + t*16 + lo) * D_MODEL + h * DK;
            FragU k0, k1;
            k0.f4[0] = *(const float4*)(krow +      hi*8);
            k0.f4[1] = *(const float4*)(krow + 16 + hi*8);
            k1.f4[0] = *(const float4*)(krow + 32 + hi*8);
            k1.f4[1] = *(const float4*)(krow + 48 + hi*8);
            s[t] = wmma_bf16(qf[0], k0, zero);
            s[t] = wmma_bf16(qf[1], k1, s[t]);
        }

        // causal mask + scale + online softmax (row = qb + r + 8*hi, key = kb + t*16 + lo)
        const int key0 = kb + lo, key1 = kb + 16 + lo;
        float p0[8], p1[8];
#pragma unroll
        for (int r = 0; r < 8; ++r) {
            const int qi = qb + r + 8*hi;
            float s0 = (key0 <= qi) ? s[0][r] * 0.125f : -1e30f;
            float s1 = (key1 <= qi) ? s[1][r] * 0.125f : -1e30f;
            float tm = fmaxf(s0, s1);
            tm = fmaxf(tm, __shfl_xor(tm, 1, 32));
            tm = fmaxf(tm, __shfl_xor(tm, 2, 32));
            tm = fmaxf(tm, __shfl_xor(tm, 4, 32));
            tm = fmaxf(tm, __shfl_xor(tm, 8, 32));
            const float mnew  = fmaxf(mr[r], tm);
            const float alpha = __expf(mr[r] - mnew);
            mr[r] = mnew;
            const float e0 = __expf(s0 - mnew);
            const float e1 = __expf(s1 - mnew);
            float rs = e0 + e1;
            rs += __shfl_xor(rs, 1, 32);
            rs += __shfl_xor(rs, 2, 32);
            rs += __shfl_xor(rs, 4, 32);
            rs += __shfl_xor(rs, 8, 32);
            lr[r] = lr[r] * alpha + rs;
#pragma unroll
            for (int j = 0; j < 4; ++j) o[j][r] *= alpha;
            p0[r] = e0; p1[r] = e1;
        }

        // stage P (16x32 bf16) to per-wave LDS in row-major [m][k], then read A-frag
        u16* pb = pbuf[wave];
#pragma unroll
        for (int r = 0; r < 8; ++r) {
            pb[(r + 8*hi) * 32 + lo]      = f2bf(p0[r]);
            pb[(r + 8*hi) * 32 + 16 + lo] = f2bf(p1[r]);
        }
        FragU pa;   // same-wave DS ops are in-order; compiler inserts s_wait_dscnt
        pa.f4[0] = *(const float4*)(pb + lo*32 +      hi*8);
        pa.f4[1] = *(const float4*)(pb + lo*32 + 16 + hi*8);

        // O += P @ V ; V pre-transposed: contraction over t contiguous
#pragma unroll
        for (int j = 0; j < 4; ++j) {
            const u16* vtrow = Vt + (size_t)(bh * DK + j*16 + lo) * TT + kb;
            FragU vb;
            vb.f4[0] = *(const float4*)(vtrow +      hi*8);
            vb.f4[1] = *(const float4*)(vtrow + 16 + hi*8);
            o[j] = wmma_bf16(pa, vb, o[j]);
        }
    }

    // normalize and store context (bf16, [b*T + q][h*64 + d])
#pragma unroll
    for (int r = 0; r < 8; ++r) {
        const float inv = 1.0f / lr[r];
        const size_t row = (size_t)(b * TT + qb + r + 8*hi) * D_MODEL + h * DK;
#pragma unroll
        for (int j = 0; j < 4; ++j)
            Ctx[row + j*16 + lo] = f2bf(o[j][r] * inv);
    }
}

// ---------- launch ----------
extern "C" void kernel_launch(void* const* d_in, const int* in_sizes, int n_in,
                              void* d_out, int out_size, void* d_ws, size_t ws_size,
                              hipStream_t stream) {
    (void)in_sizes; (void)n_in; (void)out_size; (void)ws_size;
    const float* x  = (const float*)d_in[0];
    // d_in[1] = causal mask (tril) -> implemented analytically
    const float* Wq = (const float*)d_in[2];
    const float* bq = (const float*)d_in[3];
    const float* Wk = (const float*)d_in[4];
    const float* bk = (const float*)d_in[5];
    const float* Wv = (const float*)d_in[6];
    const float* bv = (const float*)d_in[7];
    const float* Wo = (const float*)d_in[8];
    const float* bo = (const float*)d_in[9];

    const size_t SZ_X = (size_t)MROWS * D_MODEL * 2;      // 16 MB
    const size_t SZ_W = (size_t)D_MODEL * D_MODEL * 2;    // 2 MB
    char* ws = (char*)d_ws;
    u16* xb  = (u16*)(ws);                                // x bf16 (reused as ctx later)
    u16* wqb = (u16*)(ws + SZ_X);
    u16* wkb = (u16*)(ws + SZ_X + 1*SZ_W);
    u16* wvb = (u16*)(ws + SZ_X + 2*SZ_W);
    u16* wob = (u16*)(ws + SZ_X + 3*SZ_W);
    u16* Qb  = (u16*)(ws + SZ_X + 4*SZ_W);
    u16* Kb  = (u16*)(ws + 2*SZ_X + 4*SZ_W);
    u16* Vt  = (u16*)(ws + 3*SZ_X + 4*SZ_W);              // transposed V [b,h,d,t]
    u16* Ctx = xb;                                        // reuse: x dead after QKV GEMMs

    const int NX4 = MROWS * D_MODEL / 4;
    const int NW4 = D_MODEL * D_MODEL / 4;
    cvt_f32_bf16<<<NX4 / 256, 256, 0, stream>>>(x,  xb,  NX4);
    cvt_f32_bf16<<<NW4 / 256, 256, 0, stream>>>(Wq, wqb, NW4);
    cvt_f32_bf16<<<NW4 / 256, 256, 0, stream>>>(Wk, wkb, NW4);
    cvt_f32_bf16<<<NW4 / 256, 256, 0, stream>>>(Wv, wvb, NW4);
    cvt_f32_bf16<<<NW4 / 256, 256, 0, stream>>>(Wo, wob, NW4);

    dim3 ggrid(D_MODEL / 256, MROWS / 64);                // (4, 128)
    gemm_xwT<0><<<ggrid, 256, 0, stream>>>(xb, wqb, bq, (void*)Qb);
    gemm_xwT<0><<<ggrid, 256, 0, stream>>>(xb, wkb, bk, (void*)Kb);
    gemm_xwT<1><<<ggrid, 256, 0, stream>>>(xb, wvb, bv, (void*)Vt);

    const int nblocks = BB * N_HEADS * (TT / 16) / 8;     // 1024
    attn_kernel<<<nblocks, 256, 0, stream>>>(Qb, Kb, Vt, Ctx);

    gemm_xwT<2><<<ggrid, 256, 0, stream>>>(Ctx, wob, bo, d_out);
}
